// DGCNN_voxel_reshape_48026324304118
// MI455X (gfx1250) — compile-verified, matrix operand order checked by reading, not run on hardware
//
#include <hip/hip_runtime.h>
#include <hip/hip_bf16.h>
#include <math.h>

typedef __attribute__((ext_vector_type(16))) _Float16 v16h;
typedef __attribute__((ext_vector_type(8)))  _Float16 v8h;
typedef __attribute__((ext_vector_type(4)))  _Float16 v4h;
typedef __attribute__((ext_vector_type(8)))  float    v8f;

#define DIMC 256
#define HEADS 8
#define DHEAD 64
#define INNER 512
#define MLPD 1024
#define NCLS 40
#define KNNK 10
#define NPTS 200
#define MAXVOX 356
#define NB 4
#define NTOK 357
#define LREL 0.2f
#define EPSV 1e-5f

static __device__ __forceinline__ v16h cat8(v8h lo, v8h hi) {
  return __builtin_shufflevector(lo, hi, 0, 1, 2, 3, 4, 5, 6, 7, 8, 9, 10, 11, 12, 13, 14, 15);
}

// ---------------------------------------------------------------- utilities
__global__ void cast_f16_kernel(const float* __restrict__ in, _Float16* __restrict__ out, int n) {
  int i = blockIdx.x * blockDim.x + threadIdx.x;
  if (i < n) out[i] = (_Float16)in[i];
}

// W is (K x N) row-major; emit Bt = (N x K) row-major fp16 (column-contiguous B)
__global__ void transpose_cast_kernel(const float* __restrict__ W, _Float16* __restrict__ out, int K, int N) {
  int i = blockIdx.x * blockDim.x + threadIdx.x;
  if (i < K * N) {
    int k = i / N, n = i % N;
    out[(size_t)n * K + k] = (_Float16)W[i];
  }
}

__global__ void offsets_kernel(const int* __restrict__ lens, int* __restrict__ off) {
  if (threadIdx.x == 0) {
    int s = 0;
    for (int i = 0; i < NB; ++i) { off[i] = s; s += lens[i]; }
    off[NB] = s;
  }
}

__global__ void pack_kernel(const float* __restrict__ inp, const int* __restrict__ lens,
                            const int* __restrict__ off, float* __restrict__ packed) {
  int b = blockIdx.x / MAXVOX, v = blockIdx.x % MAXVOX;
  if (v >= lens[b]) return;
  int dst = off[b] + v;
  for (int t = threadIdx.x; t < NPTS * 3; t += blockDim.x)
    packed[(size_t)dst * NPTS * 3 + t] = inp[((size_t)(b * MAXVOX + v)) * NPTS * 3 + t];
}

// ---------------------------------------------------------------- kNN (dynamic graph)
__global__ void knn_kernel(const float* __restrict__ X, int xstride, int xchoff, int C,
                           int* __restrict__ idx) {
  extern __shared__ float smf[];
  float* sX  = smf;            // NPTS*C
  float* sxx = smf + NPTS * C; // NPTS
  int v = blockIdx.x, tid = threadIdx.x;
  for (int t = tid; t < NPTS * C; t += blockDim.x) {
    int n = t / C, c = t % C;
    sX[t] = X[((size_t)(v * NPTS + n)) * xstride + xchoff + c];
  }
  __syncthreads();
  for (int n = tid; n < NPTS; n += blockDim.x) {
    float s = 0.f;
    for (int c = 0; c < C; ++c) { float x = sX[n * C + c]; s += x * x; }
    sxx[n] = s;
  }
  __syncthreads();
  if (tid < NPTS) {
    int n = tid;
    float best[KNNK]; int bidx[KNNK];
    for (int q = 0; q < KNNK; ++q) { best[q] = -3.4e38f; bidx[q] = 0; }
    for (int m = 0; m < NPTS; ++m) {
      float d = 0.f;
      for (int c = 0; c < C; ++c) d += sX[n * C + c] * sX[m * C + c];
      float pd = 2.f * d - sxx[n] - sxx[m];
      pd *= (sxx[m] > 0.f) ? 1.0f : 10000000.0f;
      if (pd > best[KNNK - 1]) {
        int q = KNNK - 1;
        while (q > 0 && pd > best[q - 1]) { best[q] = best[q - 1]; bidx[q] = bidx[q - 1]; --q; }
        best[q] = pd; bidx[q] = m;
      }
    }
    for (int q = 0; q < KNNK; ++q) idx[(v * NPTS + n) * KNNK + q] = bidx[q];
  }
}

// ---------------------------------------------------------------- edge conv (gather + WMMA + max over k)
// one wave per (voxel, 16-row n-tile). W staged o-major with zero-padded K rows.
__global__ void edge_conv_kernel(const float* __restrict__ X, int xstride, int xchoff, int Cin,
                                 const _Float16* __restrict__ W, int Cout,
                                 const int* __restrict__ idx,
                                 float* __restrict__ out, int ostride, int ochoff) {
  extern __shared__ _Float16 smh[];
  const int twoC = 2 * Cin;
  const int kch  = (twoC + 31) / 32;
  const int Kpad = kch * 32;
  _Float16* sW = smh;                // Cout * Kpad (zero padded)
  _Float16* sA = smh + Cout * Kpad;  // 16 * Kpad
  int lane = threadIdx.x;
  int ntiles = (NPTS + 15) / 16;
  int v = blockIdx.x / ntiles, nt = blockIdx.x % ntiles;

  for (int t = lane; t < Cout * Kpad; t += 32) {
    int o = t / Kpad, c = t % Kpad;
    sW[t] = (c < twoC) ? W[o * twoC + c] : (_Float16)0.f;
  }
  for (int t = lane; t < 16 * Kpad; t += 32) sA[t] = (_Float16)0.f;
  __syncthreads();

  int l = lane & 15, hi = lane >> 4;
  const int otiles = Cout / 16;
  float mx[8][8];
  for (int ot = 0; ot < otiles; ++ot)
    for (int r = 0; r < 8; ++r) mx[ot][r] = -3.0e38f;

  int rr = lane >> 1, half = lane & 1;        // lane-pair per A row
  int nrow = nt * 16 + rr;
  bool rowok = (nrow < NPTS);
  size_t rown = (size_t)(v * NPTS + nrow);

  for (int k = 0; k < KNNK; ++k) {
    if (rowok) {
      int nb = idx[(v * NPTS + nrow) * KNNK + k];
      size_t rowb = (size_t)(v * NPTS + nb);
      const float* ctrp = X + rown * xstride + xchoff;
      const float* nbrp = X + rowb * xstride + xchoff;
      for (int c = half; c < Cin; c += 2) {
        float ctr = ctrp[c];
        sA[rr * Kpad + c]       = (_Float16)(nbrp[c] - ctr);  // nbr - ctr
        sA[rr * Kpad + Cin + c] = (_Float16)ctr;              // ctr
      }
    }
    __syncthreads();
    for (int ot = 0; ot < otiles; ++ot) {
      v8f acc = {};
      for (int kc = 0; kc < kch; ++kc) {
        const _Float16* ab = sA + l * Kpad + kc * 32 + hi * 8;
        const _Float16* bb = sW + (ot * 16 + l) * Kpad + kc * 32 + hi * 8;
        v16h a = cat8(*(const v8h*)ab, *(const v8h*)(ab + 16));
        v16h b = cat8(*(const v8h*)bb, *(const v8h*)(bb + 16));
        acc = __builtin_amdgcn_wmma_f32_16x16x32_f16(false, a, false, b, (short)0, acc, false, false);
      }
#pragma unroll
      for (int r = 0; r < 8; ++r) {
        float x = acc[r];
        x = x > 0.f ? x : LREL * x;            // lrelu before max, as in reference
        if (x > mx[ot][r]) mx[ot][r] = x;
      }
    }
    __syncthreads();
  }
  for (int ot = 0; ot < otiles; ++ot)
#pragma unroll
    for (int r = 0; r < 8; ++r) {
      int n = nt * 16 + r + hi * 8;
      if (n < NPTS)
        out[((size_t)(v * NPTS + n)) * ostride + ochoff + ot * 16 + l] = mx[ot][r];
    }
}

// ---------------------------------------------------------------- generic WMMA GEMM + fused epilogue
// 4 waves/block share one staged 16x32 A tile; each wave owns one 16x16 N-tile.
// Bt is (N x K) fp16 row-major (column-contiguous). N must be a multiple of 64.
__global__ void gemm_wmma_kernel(const float* __restrict__ A, int M, int K, int lda,
                                 const _Float16* __restrict__ Bt, int N,
                                 float* __restrict__ outf, _Float16* __restrict__ outh, int ldc,
                                 const float* __restrict__ bias,
                                 const float* __restrict__ gamma, const float* __restrict__ beta,
                                 const float* __restrict__ mean,  const float* __restrict__ varp,
                                 const float* __restrict__ residual, int act) {
  __shared__ _Float16 sA[16 * 32];
  int tid  = threadIdx.x;
  int lane = tid & 31, wave = tid >> 5;
  int mt = blockIdx.x;
  int nt = blockIdx.y * 4 + wave;
  int l = lane & 15, hi = lane >> 4;
  int col = nt * 16 + l;
  v8f acc = {};
  int kch = K / 32;
  int srow = tid >> 3, sseg = (tid & 7) * 4;  // 128 threads stage 16x32 floats, 4 each
  int sgr = mt * 16 + srow;
  const _Float16* bptr = Bt + (size_t)col * K + hi * 8;
  for (int kc = 0; kc < kch; ++kc) {
    v4h h = {};
    if (sgr < M) {
      const float* ap = A + (size_t)sgr * lda + kc * 32 + sseg;
#pragma unroll
      for (int e = 0; e < 4; ++e) h[e] = (_Float16)ap[e];
    }
    *(v4h*)(sA + srow * 32 + sseg) = h;
    __syncthreads();
    const _Float16* ab = sA + l * 32 + hi * 8;
    const _Float16* bb = bptr + kc * 32;
    if (kc + 1 < kch) __builtin_prefetch(bb + 32, 0, 1);
    v16h a = cat8(*(const v8h*)ab, *(const v8h*)(ab + 16));
    v16h b = cat8(*(const v8h*)bb, *(const v8h*)(bb + 16));
    acc = __builtin_amdgcn_wmma_f32_16x16x32_f16(false, a, false, b, (short)0, acc, false, false);
    __syncthreads();
  }
  float gm = 1.f, bt = 0.f, mn = 0.f, iv = 1.f;
  if (gamma) { gm = gamma[col]; bt = beta[col]; mn = mean[col]; iv = rsqrtf(varp[col] + EPSV); }
  float bs = bias ? bias[col] : 0.f;
#pragma unroll
  for (int r = 0; r < 8; ++r) {
    int row = mt * 16 + r + hi * 8;
    if (row >= M) continue;
    float v = acc[r] + bs;
    if (gamma) v = (v - mn) * iv * gm + bt;
    if (act == 1)      v = v > 0.f ? v : LREL * v;
    else if (act == 2) v = 0.5f * v * (1.f + erff(v * 0.70710678f));
    if (residual) v += residual[(size_t)row * ldc + col];
    if (outf) outf[(size_t)row * ldc + col] = v;
    else      outh[(size_t)row * ldc + col] = (_Float16)v;
  }
}

// ---------------------------------------------------------------- max+mean pooling over points
__global__ void pool_kernel(const _Float16* __restrict__ h5, float* __restrict__ g) {
  int v = blockIdx.x;
  for (int c = threadIdx.x; c < 512; c += blockDim.x) {
    float mx = -3.4e38f, sm = 0.f;
    for (int n = 0; n < NPTS; ++n) {
      float x = (float)h5[((size_t)(v * NPTS + n)) * 512 + c];
      mx = fmaxf(mx, x); sm += x;
    }
    g[(size_t)v * 1024 + c]       = mx;
    g[(size_t)v * 1024 + 512 + c] = sm * (1.f / NPTS);
  }
}

// ---------------------------------------------------------------- token build (cls + scatter + PE)
__global__ void tokens_kernel(const float* __restrict__ vf, const int* __restrict__ lens,
                              const int* __restrict__ off, const float* __restrict__ vseq,
                              const float* __restrict__ cls_token, float* __restrict__ x) {
  int b = blockIdx.x / NTOK, t = blockIdx.x % NTOK;
  float* row = x + ((size_t)(b * NTOK + t)) * DIMC;
  if (t == 0) {
    for (int c = threadIdx.x; c < DIMC; c += blockDim.x) row[c] = cls_token[c];
    return;
  }
  int v = t - 1, lb = lens[b];
  float pos = vseq[b * MAXVOX + v];
  for (int c = threadIdx.x; c < DIMC; c += blockDim.x) {
    float val = 0.f;
    if (v < lb) val = vf[((size_t)(off[b] + v)) * DIMC + c];
    if (c < lb) {  // reference masks PE by channel-index < len
      int i = c >> 1;
      float div = __expf(-(2.0f * i) * (9.210340371976184f / 256.0f));
      float p = pos * div;
      val += (c & 1) ? __cosf(p) : __sinf(p);
    }
    row[c] = val;
  }
}

// ---------------------------------------------------------------- layer norm (256 threads = DIM)
__global__ void ln_kernel(const float* __restrict__ x, const float* __restrict__ g,
                          const float* __restrict__ b, float* __restrict__ out) {
  __shared__ float red[DIMC];
  int t = blockIdx.x, tid = threadIdx.x;
  float v = x[(size_t)t * DIMC + tid];
  red[tid] = v; __syncthreads();
  for (int s = DIMC / 2; s > 0; s >>= 1) { if (tid < s) red[tid] += red[tid + s]; __syncthreads(); }
  float mean = red[0] / DIMC; __syncthreads();
  float d = v - mean;
  red[tid] = d * d; __syncthreads();
  for (int s = DIMC / 2; s > 0; s >>= 1) { if (tid < s) red[tid] += red[tid + s]; __syncthreads(); }
  float var = red[0] / DIMC;
  out[(size_t)t * DIMC + tid] = d * rsqrtf(var + EPSV) * g[tid] + b[tid];
}

// ---------------------------------------------------------------- softmax attention, one block per (b,h)
__global__ void attn_kernel(const float* __restrict__ qkv, float* __restrict__ out) {
  extern __shared__ float sma[];
  float* sc  = sma;          // NTOK
  float* red = sma + NTOK;   // blockDim
  int b = blockIdx.x / HEADS, h = blockIdx.x % HEADS;
  int tid = threadIdx.x;     // 128
  const float scale = 0.125f;
  for (int i = 0; i < NTOK; ++i) {
    const float* q = qkv + ((size_t)(b * NTOK + i)) * 1536 + h * DHEAD;
    for (int j = tid; j < NTOK; j += 128) {
      const float* kp = qkv + ((size_t)(b * NTOK + j)) * 1536 + 512 + h * DHEAD;
      float s = 0.f;
      for (int d = 0; d < DHEAD; ++d) s += q[d] * kp[d];
      sc[j] = s * scale;
    }
    __syncthreads();
    float m = -3.4e38f;
    for (int j = tid; j < NTOK; j += 128) m = fmaxf(m, sc[j]);
    red[tid] = m; __syncthreads();
    for (int s = 64; s > 0; s >>= 1) { if (tid < s) red[tid] = fmaxf(red[tid], red[tid + s]); __syncthreads(); }
    float mx = red[0]; __syncthreads();
    float ps = 0.f;
    for (int j = tid; j < NTOK; j += 128) { float e = __expf(sc[j] - mx); sc[j] = e; ps += e; }
    red[tid] = ps; __syncthreads();
    for (int s = 64; s > 0; s >>= 1) { if (tid < s) red[tid] += red[tid + s]; __syncthreads(); }
    float inv = 1.f / red[0]; __syncthreads();
    if (tid < DHEAD) {
      float a = 0.f;
      for (int j = 0; j < NTOK; ++j)
        a += sc[j] * qkv[((size_t)(b * NTOK + j)) * 1536 + 1024 + h * DHEAD + tid];
      out[((size_t)(b * NTOK + i)) * INNER + h * DHEAD + tid] = a * inv;
    }
    __syncthreads();
  }
}

// ---------------------------------------------------------------- classifier head (tiny)
__global__ void classifier_kernel(const float* __restrict__ x, const float* __restrict__ C1,
                                  const float* __restrict__ C1b, const float* __restrict__ C2,
                                  const float* __restrict__ C2b, float* __restrict__ logits) {
  __shared__ float hid[512];
  int b = blockIdx.x, tid = threadIdx.x;
  const float* cls = x + (size_t)(b * NTOK) * DIMC;
  for (int o = tid; o < 512; o += blockDim.x) {
    float s = C1b[o];
    for (int c = 0; c < DIMC; ++c) s += cls[c] * C1[c * 512 + o];
    hid[o] = s > 0.f ? s : LREL * s;
  }
  __syncthreads();
  for (int o = tid; o < NCLS; o += blockDim.x) {
    float s = C2b[o];
    for (int c = 0; c < 512; ++c) s += hid[c] * C2[c * NCLS + o];
    logits[b * NCLS + o] = s;
  }
}

// ================================================================ host
extern "C" void kernel_launch(void* const* d_in, const int* in_sizes, int n_in,
                              void* d_out, int out_size, void* d_ws, size_t ws_size,
                              hipStream_t stream) {
  const int V = (out_size - NB * NCLS) / DIMC;   // sum(lens), derived from output layout
  const int Btok = NB * NTOK;

  const float* inp  = (const float*)d_in[0];
  const int*   lens = (const int*)d_in[1];
  const float* vseq = (const float*)d_in[2];
  // params flattened in sorted-key (jax pytree) order
  int p = 3;
  const float* C1  = (const float*)d_in[p++];
  const float* C1b = (const float*)d_in[p++];
  const float* C2  = (const float*)d_in[p++];
  const float* C2b = (const float*)d_in[p++];
  const float* cls_token = (const float*)d_in[p++];
  const float* L1  = (const float*)d_in[p++];
  const float* L2  = (const float*)d_in[p++];
  const float* L2b = (const float*)d_in[p++];
  const float* W1  = (const float*)d_in[p++];
  const float* W2  = (const float*)d_in[p++];
  const float* W3  = (const float*)d_in[p++];
  const float* W4  = (const float*)d_in[p++];
  const float* W5  = (const float*)d_in[p++];
  const float *bn5b = (const float*)d_in[p++], *bn5g = (const float*)d_in[p++],
              *bn5m = (const float*)d_in[p++], *bn5v = (const float*)d_in[p++];
  const float *bn6b = (const float*)d_in[p++], *bn6g = (const float*)d_in[p++],
              *bn6m = (const float*)d_in[p++], *bn6v = (const float*)d_in[p++];
  const float *bn7b = (const float*)d_in[p++], *bn7g = (const float*)d_in[p++],
              *bn7m = (const float*)d_in[p++], *bn7v = (const float*)d_in[p++];
  const float *F1w[4], *F1b[4], *F2w[4], *F2b[4], *Wout[4], *Woutb[4], *Wqkv[4],
              *ln1b[4], *ln1g[4], *ln2b[4], *ln2g[4];
  for (int l = 0; l < 4; ++l) {
    F1w[l]  = (const float*)d_in[p++]; F1b[l]  = (const float*)d_in[p++];
    F2w[l]  = (const float*)d_in[p++]; F2b[l]  = (const float*)d_in[p++];
    Wout[l] = (const float*)d_in[p++]; Woutb[l] = (const float*)d_in[p++];
    Wqkv[l] = (const float*)d_in[p++];
    ln1b[l] = (const float*)d_in[p++]; ln1g[l] = (const float*)d_in[p++];
    ln2b[l] = (const float*)d_in[p++]; ln2g[l] = (const float*)d_in[p++];
  }

  // workspace carve-out
  char* wsb = (char*)d_ws; size_t pos = 0;
  auto alloc = [&](size_t bytes) -> void* {
    void* r = wsb + pos; pos = (pos + bytes + 255) & ~(size_t)255; return r;
  };
  int*       off    = (int*)alloc((NB + 1) * 4);
  float*     packed = (float*)alloc((size_t)V * NPTS * 3 * 4);
  int*       idx    = (int*)alloc((size_t)V * NPTS * KNNK * 4);
  float*     hcat   = (float*)alloc((size_t)V * NPTS * 256 * 4);
  _Float16*  h5     = (_Float16*)alloc((size_t)V * NPTS * 512 * 2);
  float*     g      = (float*)alloc((size_t)V * 1024 * 4);
  float*     g1     = (float*)alloc((size_t)V * 1024 * 4);
  float*     xtok   = (float*)alloc((size_t)Btok * DIMC * 4);
  float*     xn     = (float*)alloc((size_t)Btok * DIMC * 4);
  float*     qkvb   = (float*)alloc((size_t)Btok * 1536 * 4);
  float*     attb   = (float*)alloc((size_t)Btok * INNER * 4);
  float*     ffh    = (float*)alloc((size_t)Btok * MLPD * 4);
  _Float16*  W1h    = (_Float16*)alloc(32 * 6 * 2);
  _Float16*  W2h    = (_Float16*)alloc(32 * 64 * 2);
  _Float16*  W3h    = (_Float16*)alloc(64 * 64 * 2);
  _Float16*  W4h    = (_Float16*)alloc(128 * 128 * 2);
  _Float16*  W5t    = (_Float16*)alloc((size_t)512 * 256 * 2);   // already (N x K)
  _Float16*  L1t    = (_Float16*)alloc((size_t)1024 * 1024 * 2);
  _Float16*  L2t    = (_Float16*)alloc((size_t)256 * 1024 * 2);
  _Float16 *qkvt[4], *woutt[4], *f1t[4], *f2t[4];
  for (int l = 0; l < 4; ++l) {
    qkvt[l]  = (_Float16*)alloc((size_t)1536 * 256 * 2);
    woutt[l] = (_Float16*)alloc((size_t)256 * 512 * 2);
    f1t[l]   = (_Float16*)alloc((size_t)1024 * 256 * 2);
    f2t[l]   = (_Float16*)alloc((size_t)256 * 1024 * 2);
  }
  (void)ws_size; (void)n_in; (void)in_sizes;

  float* logits = (float*)d_out;
  float* vf     = (float*)d_out + NB * NCLS;

  auto cast = [&](const float* src, _Float16* dst, int n) {
    cast_f16_kernel<<<(n + 255) / 256, 256, 0, stream>>>(src, dst, n);
  };
  auto tcast = [&](const float* src, _Float16* dst, int K, int N) {
    transpose_cast_kernel<<<(K * N + 255) / 256, 256, 0, stream>>>(src, dst, K, N);
  };
  cast(W1, W1h, 32 * 6); cast(W2, W2h, 32 * 64); cast(W3, W3h, 64 * 64); cast(W4, W4h, 128 * 128);
  cast(W5, W5t, 512 * 256);                // (O x C) == (N x K) already
  tcast(L1, L1t, 1024, 1024);
  tcast(L2, L2t, 1024, 256);
  for (int l = 0; l < 4; ++l) {
    tcast(Wqkv[l], qkvt[l], 256, 1536);
    tcast(Wout[l], woutt[l], 512, 256);
    tcast(F1w[l], f1t[l], 256, 1024);
    tcast(F2w[l], f2t[l], 1024, 256);
  }

  offsets_kernel<<<1, 32, 0, stream>>>(lens, off);
  pack_kernel<<<NB * MAXVOX, 128, 0, stream>>>(inp, lens, off, packed);

  const int ntiles = (NPTS + 15) / 16;
  auto knn = [&](const float* X, int stride, int choff, int C) {
    knn_kernel<<<V, 256, (size_t)(NPTS * C + NPTS) * 4, stream>>>(X, stride, choff, C, idx);
  };
  auto econv = [&](const float* X, int stride, int choff, int Cin, const _Float16* Wh, int Cout, int ochoff) {
    int Kpad = ((2 * Cin + 31) / 32) * 32;
    size_t sh = (size_t)(Cout * Kpad + 16 * Kpad) * 2;
    edge_conv_kernel<<<V * ntiles, 32, sh, stream>>>(X, stride, choff, Cin, Wh, Cout, idx, hcat, 256, ochoff);
  };
  knn(packed, 3, 0, 3);    econv(packed, 3, 0, 3,   W1h,  32,   0);
  knn(hcat, 256, 0, 32);   econv(hcat, 256, 0, 32,  W2h,  32,  32);
  knn(hcat, 256, 32, 32);  econv(hcat, 256, 32, 32, W3h,  64,  64);
  knn(hcat, 256, 64, 64);  econv(hcat, 256, 64, 64, W4h, 128, 128);

  auto gemm = [&](const float* A, int M, int K, int lda, const _Float16* Bt, int N,
                  float* of, _Float16* oh, int ldc, const float* bias,
                  const float* bg, const float* bb, const float* bm, const float* bv,
                  const float* res, int act) {
    dim3 grid((M + 15) / 16, N / 64);
    gemm_wmma_kernel<<<grid, 128, 0, stream>>>(A, M, K, lda, Bt, N, of, oh, ldc,
                                               bias, bg, bb, bm, bv, res, act);
  };

  // h5 = lrelu(bn5(hcat @ W5^T))  (fp16)
  gemm(hcat, V * NPTS, 256, 256, W5t, 512, nullptr, h5, 512,
       nullptr, bn5g, bn5b, bn5m, bn5v, nullptr, 1);
  pool_kernel<<<V, 256, 0, stream>>>(h5, g);
  // g1 = lrelu(bn6(g @ L1))
  gemm(g, V, 1024, 1024, L1t, 1024, g1, nullptr, 1024,
       nullptr, bn6g, bn6b, bn6m, bn6v, nullptr, 1);
  // vf = lrelu(bn7(g1 @ L2 + L2b)) -> straight into d_out
  gemm(g1, V, 1024, 1024, L2t, 256, vf, nullptr, 256,
       L2b, bn7g, bn7b, bn7m, bn7v, nullptr, 1);

  tokens_kernel<<<NB * NTOK, 128, 0, stream>>>(vf, lens, off, vseq, cls_token, xtok);

  for (int l = 0; l < 4; ++l) {
    ln_kernel<<<Btok, DIMC, 0, stream>>>(xtok, ln1g[l], ln1b[l], xn);
    gemm(xn, Btok, 256, 256, qkvt[l], 1536, qkvb, nullptr, 1536,
         nullptr, nullptr, nullptr, nullptr, nullptr, nullptr, 0);
    attn_kernel<<<NB * HEADS, 128, (size_t)(NTOK + 128) * 4, stream>>>(qkvb, attb);
    gemm(attb, Btok, 512, 512, woutt[l], 256, xtok, nullptr, 256,
         Woutb[l], nullptr, nullptr, nullptr, nullptr, xtok, 0);
    ln_kernel<<<Btok, DIMC, 0, stream>>>(xtok, ln2g[l], ln2b[l], xn);
    gemm(xn, Btok, 256, 256, f1t[l], 1024, ffh, nullptr, 1024,
         F1b[l], nullptr, nullptr, nullptr, nullptr, nullptr, 2);
    gemm(ffh, Btok, 1024, 1024, f2t[l], 256, xtok, nullptr, 256,
         F2b[l], nullptr, nullptr, nullptr, nullptr, xtok, 0);
  }

  classifier_kernel<<<NB, 128, 0, stream>>>(xtok, C1, C1b, C2, C2b, logits);
}
